// ActorTorsionNet_74543452389454
// MI455X (gfx1250) — matrix operand order repaired and checked by reading, not compiled, not run on hardware
//
#include <hip/hip_runtime.h>

#define NNODES 20000
#define NEDGES 60000
#define NTORS  1024
#define NTILE_E (NEDGES/16)   // 3750
#define NTILE_N (NNODES/16)   // 1250
#define WV 4                  // waves per block (wave32)

typedef __attribute__((ext_vector_type(16))) _Float16 v16h;
typedef __attribute__((ext_vector_type(8)))  float    v8f;

__device__ __forceinline__ v8f wmma_f16(v16h a, v16h b, v8f c){
  return __builtin_amdgcn_wmma_f32_16x16x32_f16(false, a, false, b, (short)0, c, false, false);
}
__device__ __forceinline__ int kmapA(int v, int hi){ return 16*(v>>2) + 8*hi + 2*(v&3); }
__device__ __forceinline__ int kmapB(int v, int hi){ return 16*(v>>2) + 8*hi + 2*(v&3); }

// A[m,k] = base[m*ld + k], rows = lane&15, K-step s (32 wide)
__device__ __forceinline__ v16h load_A16(const _Float16* base, int ld, int s){
  const int lane = threadIdx.x & 31, m = lane & 15, hi = lane >> 4;
  v16h A;
  #pragma unroll
  for (int v = 0; v < 8; ++v){
    const int k = s*32 + kmapA(v, hi);
    const _Float16* p = base + (size_t)m*ld + k;
    A[2*v] = p[0]; A[2*v+1] = p[1];
  }
  return A;
}
// B[k,n] = W[(col0+n)*ldk + k]   (i.e. X @ W^T with W row-major [out,in])
__device__ __forceinline__ v16h load_B_T(const _Float16* W, int col0, int ldk, int s){
  const int lane = threadIdx.x & 31, n = lane & 15, hi = lane >> 4;
  v16h B;
  #pragma unroll
  for (int v = 0; v < 8; ++v){
    const int k = s*32 + kmapB(v, hi);
    const _Float16* p = W + (size_t)(col0+n)*ldk + k;
    B[2*v] = p[0]; B[2*v+1] = p[1];
  }
  return B;
}
// B[k,n] = M[k*ldn + col0+n]     (i.e. X @ M, M row-major [in,out])
__device__ __forceinline__ v16h load_B_RM(const _Float16* M, int col0, int ldn, int s){
  const int lane = threadIdx.x & 31, n = lane & 15, hi = lane >> 4;
  v16h B;
  #pragma unroll
  for (int v = 0; v < 8; ++v){
    const int k = s*32 + kmapB(v, hi);
    B[2*v]   = M[(size_t)k*ldn + col0 + n];
    B[2*v+1] = M[(size_t)(k+1)*ldn + col0 + n];
  }
  return B;
}
__device__ __forceinline__ v8f splat8(float v){
  v8f c;
  #pragma unroll
  for (int r = 0; r < 8; ++r) c[r] = v;
  return c;
}
__device__ __forceinline__ float sigm(float x){ return 1.f/(1.f + __expf(-x)); }
__device__ __forceinline__ unsigned fkey(float f){
  unsigned u = __float_as_uint(f);
  return (u & 0x80000000u) ? ~u : (u | 0x80000000u);
}
__device__ __forceinline__ float fkey_inv(unsigned k){
  unsigned u = (k & 0x80000000u) ? (k & 0x7fffffffu) : ~k;
  return __uint_as_float(u);
}

// ---------------- utility kernels ----------------
__global__ void fzero_kernel(float* __restrict__ p, int n){
  const int i = blockIdx.x*blockDim.x + threadIdx.x;
  if (i < n) p[i] = 0.f;
}
__global__ void cvt_f16_kernel(const float* __restrict__ s, _Float16* __restrict__ d, int n){
  const int i = blockIdx.x*blockDim.x + threadIdx.x;
  if (i < n) d[i] = (_Float16)s[i];
}
__global__ void deg_kernel(const int* __restrict__ ei, float* __restrict__ deg){
  const int i = blockIdx.x*blockDim.x + threadIdx.x;
  if (i < NEDGES) atomicAdd(deg + ei[NEDGES + i], 1.f);
}
__global__ void invdeg_kernel(float* __restrict__ d){
  const int i = blockIdx.x*blockDim.x + threadIdx.x;
  if (i < NNODES) d[i] = 1.f/fmaxf(d[i], 1.f);
}

// ---------------- front end ----------------
__global__ void lin0_kernel(const float* __restrict__ x, const float* __restrict__ W0,
                            const float* __restrict__ b0, float* __restrict__ out,
                            _Float16* __restrict__ o16){
  const int i = blockIdx.x*blockDim.x + threadIdx.x;
  if (i >= NNODES*64) return;
  const int n = i >> 6, f = i & 63;
  float v = b0[f] + x[n*3]*W0[f*3] + x[n*3+1]*W0[f*3+1] + x[n*3+2]*W0[f*3+2];
  v = fmaxf(v, 0.f);
  out[i] = v; o16[i] = (_Float16)v;
}
__global__ void edge_hidden_kernel(const float* __restrict__ ea, const float* __restrict__ We1,
                                   const float* __restrict__ be1, float* __restrict__ hidden){
  const int i = blockIdx.x*blockDim.x + threadIdx.x;
  if (i >= NEDGES*128) return;
  const int e = i >> 7, k = i & 127;
  float acc = be1[k];
  #pragma unroll
  for (int j = 0; j < 7; ++j) acc += ea[e*7 + j]*We1[k*7 + j];
  hidden[i] = fmaxf(acc, 0.f);
}
// pack hidden [E,128] into WMMA A-operand layout: [tile][s][lane][16 halves]
__global__ void swz_hidden_kernel(const float* __restrict__ hidden, _Float16* __restrict__ hsw){
  const size_t idx = (size_t)blockIdx.x*blockDim.x + threadIdx.x;
  if (idx >= (size_t)NEDGES*128) return;
  const int h = (int)(idx & 15); const size_t r = idx >> 4;
  const int lane = (int)(r & 31); const size_t r2 = r >> 5;
  const int s = (int)(r2 & 3); const int tile = (int)(r2 >> 2);
  const int p = h & 1, v = h >> 1;
  const int m = lane & 15, hi = lane >> 4;
  const int k = s*32 + kmapA(v, hi) + p;
  hsw[idx] = (_Float16)hidden[(size_t)(tile*16 + m)*128 + k];
}
// pack We2 [4096,128] into WMMA B-operand layout (B = We2^T): [j][s][lane][16 halves]
__global__ void swz_we2_kernel(const float* __restrict__ We2, _Float16* __restrict__ wsw){
  const size_t idx = (size_t)blockIdx.x*blockDim.x + threadIdx.x;
  if (idx >= (size_t)4096*128) return;
  const int h = (int)(idx & 15); const size_t r = idx >> 4;
  const int lane = (int)(r & 31); const size_t r2 = r >> 5;
  const int s = (int)(r2 & 3); const int j = (int)(r2 >> 2);
  const int p = h & 1, v = h >> 1;
  const int n = lane & 15, hi = lane >> 4;
  const int k = s*32 + kmapB(v, hi) + p;
  wsw[idx] = (_Float16)We2[(size_t)(j*16 + n)*128 + k];
}

// ---------------- fused NNConv message pass ----------------
// msg[e,f] = sum_d out[src,d] * (hidden[e,:]@We2[d*64+f,:] + be2[d*64+f]) -> atomicAdd agg[dst]
__global__ void msg_kernel(const float* __restrict__ out,
                           const _Float16* __restrict__ hid_sw,
                           const _Float16* __restrict__ we2_sw,
                           const float* __restrict__ be2,
                           const int* __restrict__ edge_index,
                           float* __restrict__ agg){
  __shared__ float s_be2[4096];
  __shared__ float s_out[WV][16*64];
  const int tid = threadIdx.x;
  const int lane = tid & 31;
  const int wv = tid >> 5;
  const int nn = lane & 15;
  const int hi8 = (lane >> 4)*8;
  const int tile = blockIdx.x*WV + wv;
  const bool act = (tile < NTILE_E);

  for (int i = tid; i < 4096; i += blockDim.x) s_be2[i] = be2[i];

  int dstn[8];
  v16h A0{}, A1{}, A2{}, A3{};
  if (act){
    const int e0 = tile*16;
    for (int i = lane; i < 256; i += 32){           // gather out[src] rows (16 x 64 f32)
      const int m = i >> 4, c4 = i & 15;
      const int srcn = edge_index[e0 + m];
      const float4 v = *(const float4*)(out + (size_t)srcn*64 + c4*4);
      *(float4*)(&s_out[wv][m*64 + c4*4]) = v;
    }
    #pragma unroll
    for (int r = 0; r < 8; ++r) dstn[r] = edge_index[NEDGES + e0 + r + hi8];
    const _Float16* ap = hid_sw + (size_t)tile*2048 + (size_t)lane*16;
    A0 = *(const v16h*)(ap);
    A1 = *(const v16h*)(ap + 512);
    A2 = *(const v16h*)(ap + 1024);
    A3 = *(const v16h*)(ap + 1536);
  }
  __syncthreads();
  if (!act) return;

  v8f macc[4] = {};
  const float* so = &s_out[wv][0];
  for (int jd = 0; jd < 64; ++jd){                  // d dimension
    float sc[8];
    #pragma unroll
    for (int r = 0; r < 8; ++r) sc[r] = so[(r + hi8)*64 + jd];
    #pragma unroll
    for (int fb = 0; fb < 4; ++fb){                 // f blocks of 16
      const int j = jd*4 + fb;
      const _Float16* bp = we2_sw + (size_t)j*2048 + (size_t)lane*16;
      v16h B0 = *(const v16h*)(bp);
      v16h B1 = *(const v16h*)(bp + 512);
      v16h B2 = *(const v16h*)(bp + 1024);
      v16h B3 = *(const v16h*)(bp + 1536);
      v8f c = splat8(s_be2[j*16 + nn]);             // fold be2 bias into C
      c = wmma_f16(A0, B0, c);
      c = wmma_f16(A1, B1, c);
      c = wmma_f16(A2, B2, c);
      c = wmma_f16(A3, B3, c);
      #pragma unroll
      for (int r = 0; r < 8; ++r) macc[fb][r] += c[r]*sc[r];
    }
    if (jd < 63)
      __builtin_prefetch(we2_sw + (size_t)(jd + 1)*4*2048 + (size_t)lane*16, 0, 0);
  }
  #pragma unroll
  for (int fb = 0; fb < 4; ++fb){
    #pragma unroll
    for (int r = 0; r < 8; ++r)
      atomicAdd(agg + (size_t)dstn[r]*64 + fb*16 + nn, macc[fb][r]);
  }
}

// ---------------- node update: m = relu(agg/deg + out@root + b); GRU ----------------
__global__ void node_update_kernel(const float* __restrict__ agg, const float* __restrict__ invdeg,
    const float* __restrict__ out_old, const _Float16* __restrict__ o16_old,
    const _Float16* __restrict__ root16, const _Float16* __restrict__ wih16,
    const _Float16* __restrict__ whh16,
    const float* __restrict__ conv_b, const float* __restrict__ bih, const float* __restrict__ bhh,
    float* __restrict__ out_new, _Float16* __restrict__ o16_new){
  __shared__ _Float16 s_m[WV][16*64];
  const int lane = threadIdx.x & 31, wv = threadIdx.x >> 5;
  const int tile = blockIdx.x*WV + wv;
  if (tile >= NTILE_N) return;
  const int n0 = tile*16;
  const int nn = lane & 15, hi8 = (lane >> 4)*8;

  const v16h Ah0 = load_A16(o16_old + (size_t)n0*64, 64, 0);
  const v16h Ah1 = load_A16(o16_old + (size_t)n0*64, 64, 1);

  #pragma unroll
  for (int jn = 0; jn < 4; ++jn){                   // m = relu(agg*invdeg + out@root + conv_b)
    const int col = jn*16 + nn;
    v8f c = splat8(conv_b[col]);
    c = wmma_f16(Ah0, load_B_RM(root16, jn*16, 64, 0), c);
    c = wmma_f16(Ah1, load_B_RM(root16, jn*16, 64, 1), c);
    #pragma unroll
    for (int r = 0; r < 8; ++r){
      const int row = n0 + r + hi8;
      const float v = c[r] + agg[(size_t)row*64 + col]*invdeg[row];
      s_m[wv][(r + hi8)*64 + col] = (_Float16)fmaxf(v, 0.f);
    }
  }
  const v16h Am0 = load_A16(&s_m[wv][0], 64, 0);
  const v16h Am1 = load_A16(&s_m[wv][0], 64, 1);

  #pragma unroll
  for (int jr = 0; jr < 4; ++jr){                   // GRU gates r,z,n
    const int col = jr*16 + nn;
    v8f gr = splat8(bih[col]),       hr = splat8(bhh[col]);
    v8f gz = splat8(bih[64 + col]),  hz = splat8(bhh[64 + col]);
    v8f gn = splat8(bih[128 + col]), hn = splat8(bhh[128 + col]);
    gr = wmma_f16(Am0, load_B_T(wih16, jr*16,        64, 0), gr);
    gz = wmma_f16(Am0, load_B_T(wih16, 64 + jr*16,   64, 0), gz);
    gn = wmma_f16(Am0, load_B_T(wih16, 128 + jr*16,  64, 0), gn);
    hr = wmma_f16(Ah0, load_B_T(whh16, jr*16,        64, 0), hr);
    hz = wmma_f16(Ah0, load_B_T(whh16, 64 + jr*16,   64, 0), hz);
    hn = wmma_f16(Ah0, load_B_T(whh16, 128 + jr*16,  64, 0), hn);
    gr = wmma_f16(Am1, load_B_T(wih16, jr*16,        64, 1), gr);
    gz = wmma_f16(Am1, load_B_T(wih16, 64 + jr*16,   64, 1), gz);
    gn = wmma_f16(Am1, load_B_T(wih16, 128 + jr*16,  64, 1), gn);
    hr = wmma_f16(Ah1, load_B_T(whh16, jr*16,        64, 1), hr);
    hz = wmma_f16(Ah1, load_B_T(whh16, 64 + jr*16,   64, 1), hz);
    hn = wmma_f16(Ah1, load_B_T(whh16, 128 + jr*16,  64, 1), hn);
    #pragma unroll
    for (int r = 0; r < 8; ++r){
      const int row = n0 + r + hi8;
      const float rg = sigm(gr[r] + hr[r]);
      const float zg = sigm(gz[r] + hz[r]);
      const float ng = tanhf(gn[r] + rg*hn[r]);
      const float hold = out_old[(size_t)row*64 + col];
      const float hnew = (1.f - zg)*ng + zg*hold;
      out_new[(size_t)row*64 + col] = hnew;
      o16_new[(size_t)row*64 + col] = (_Float16)hnew;
    }
  }
}

// ---------------- Set2Set ----------------
// sm layout: [0,128) q_star | [128,192) hs | [192,256) cs | [256,320) rvec acc | [320] sum | [321] maxkey
__global__ void s2s_lstm_kernel(const float* __restrict__ wi, const float* __restrict__ wh,
                                const float* __restrict__ bi, const float* __restrict__ bh,
                                float* __restrict__ sm){
  __shared__ float qst[128], hso[64], cso[64], g[256];
  const int t = threadIdx.x;
  if (t < 128) qst[t] = sm[t];
  if (t < 64){ hso[t] = sm[128 + t]; cso[t] = sm[192 + t]; }
  __syncthreads();
  float acc = bi[t] + bh[t];
  for (int j = 0; j < 128; ++j) acc += qst[j]*wi[t*128 + j];
  for (int j = 0; j < 64;  ++j) acc += hso[j]*wh[t*64 + j];
  g[t] = acc;
  __syncthreads();
  if (t < 64){
    const float ig = sigm(g[t]);
    const float fg = sigm(g[64 + t]);
    const float gg = tanhf(g[128 + t]);
    const float og = sigm(g[192 + t]);
    const float cs = fg*cso[t] + ig*gg;
    sm[192 + t] = cs;
    sm[128 + t] = og*tanhf(cs);
    sm[256 + t] = 0.f;
  }
  if (t == 0){ sm[320] = 0.f; sm[321] = __uint_as_float(0u); }
}
__global__ void s2s_scores_kernel(const float* __restrict__ out, float* __restrict__ evec,
                                  float* __restrict__ sm){
  __shared__ float q[64];
  if (threadIdx.x < 64) q[threadIdx.x] = sm[128 + threadIdx.x];
  __syncthreads();
  const int n = blockIdx.x*blockDim.x + threadIdx.x;
  if (n >= NNODES) return;
  float acc = 0.f;
  for (int k = 0; k < 64; ++k) acc += out[(size_t)n*64 + k]*q[k];
  evec[n] = acc;
  atomicMax((unsigned*)(sm + 321), fkey(acc));
}
__global__ void s2s_accum_kernel(const float* __restrict__ out, const float* __restrict__ evec,
                                 float* __restrict__ sm){
  __shared__ float sw[256];
  const int t = threadIdx.x;
  const int n0 = blockIdx.x*256;
  const int n = n0 + t;
  const float mx = fkey_inv(__float_as_uint(sm[321]));
  float w = 0.f;
  if (n < NNODES) w = __expf(evec[n] - mx);
  sw[t] = w;
  __syncthreads();
  const int lim = (NNODES - n0 < 256) ? (NNODES - n0) : 256;
  if (t < 64){
    float acc = 0.f;
    for (int i = 0; i < lim; ++i) acc += sw[i]*out[(size_t)(n0 + i)*64 + t];
    atomicAdd(sm + 256 + t, acc);
  }
  if (t == 0){
    float s = 0.f;
    for (int i = 0; i < lim; ++i) s += sw[i];
    atomicAdd(sm + 320, s);
  }
}
__global__ void s2s_final_kernel(float* __restrict__ sm){
  const int t = threadIdx.x;
  if (t < 64) sm[t] = sm[128 + t];
  else if (t < 128) sm[t] = sm[256 + (t - 64)]/sm[320];
}

// ---------------- torsion head ----------------
// z_in[t,a] = out[nonring_flat[a*16 + t/64]][t%64] (a<256) ; pool[(t*128+c)/1024] (a=256+c)
__global__ void build_zin_kernel(const float* __restrict__ out, const int* __restrict__ nonring,
                                 const float* __restrict__ sm, _Float16* __restrict__ zin){
  const int idx = blockIdx.x*blockDim.x + threadIdx.x;
  if (idx >= NTORS*384) return;
  const int t = idx/384, a = idx%384;
  float v;
  if (a < 256){
    const int node = nonring[a*16 + (t >> 6)];
    v = out[(size_t)node*64 + (t & 63)];
  } else {
    v = sm[(t*128 + (a - 256)) >> 10];
  }
  zin[(size_t)t*384 + a] = (_Float16)v;
}
// memory LSTM (h0=c0=0): g = zin@mem_wi^T + bi + bh ; hm = sig(o)*tanh(sig(i)*tanh(g))
__global__ void mem_lstm_kernel(const _Float16* __restrict__ zin, const _Float16* __restrict__ wi16,
                                const float* __restrict__ bi, const float* __restrict__ bh,
                                _Float16* __restrict__ hm){
  const int lane = threadIdx.x & 31, wv = threadIdx.x >> 5;
  const int wid = blockIdx.x*WV + wv;
  if (wid >= 64*24) return;
  const int mt = wid/24, jn = wid%24;
  const int nn = lane & 15, hi8 = (lane >> 4)*8;
  const int col = jn*16 + nn;
  v8f ci = splat8(bi[col] + bh[col]);
  v8f cg = splat8(bi[768 + col] + bh[768 + col]);
  v8f co = splat8(bi[1152 + col] + bh[1152 + col]);
  const _Float16* abase = zin + (size_t)mt*16*384;
  for (int s = 0; s < 12; ++s){
    const v16h A = load_A16(abase, 384, s);
    ci = wmma_f16(A, load_B_T(wi16, jn*16,        384, s), ci);
    cg = wmma_f16(A, load_B_T(wi16, 768 + jn*16,  384, s), cg);
    co = wmma_f16(A, load_B_T(wi16, 1152 + jn*16, 384, s), co);
  }
  #pragma unroll
  for (int r = 0; r < 8; ++r){
    const float h = sigm(co[r])*tanhf(sigm(ci[r])*tanhf(cg[r]));
    const int row = mt*16 + r + hi8;
    hm[(size_t)row*384 + col] = (_Float16)h;
  }
}
__global__ void head_u_kernel(const _Float16* __restrict__ hm, const _Float16* __restrict__ w116,
                              const float* __restrict__ b1, _Float16* __restrict__ u16){
  const int lane = threadIdx.x & 31, wv = threadIdx.x >> 5;
  const int wid = blockIdx.x*WV + wv;
  if (wid >= 64*8) return;
  const int mt = wid >> 3, jn = wid & 7;
  const int nn = lane & 15, hi8 = (lane >> 4)*8;
  const int col = jn*16 + nn;
  v8f c = splat8(b1[col]);
  const _Float16* abase = hm + (size_t)mt*16*384;
  for (int s = 0; s < 12; ++s){
    const v16h A = load_A16(abase, 384, s);
    c = wmma_f16(A, load_B_T(w116, jn*16, 384, s), c);
  }
  #pragma unroll
  for (int r = 0; r < 8; ++r){
    const int row = mt*16 + r + hi8;
    u16[(size_t)row*128 + col] = (_Float16)fmaxf(c[r], 0.f);
  }
}
__global__ void head_y_kernel(const _Float16* __restrict__ u16, const float* __restrict__ W2,
                              const float* __restrict__ b2, float* __restrict__ y){
  const int idx = blockIdx.x*blockDim.x + threadIdx.x;
  if (idx >= NTORS*6) return;
  const int t = idx/6, a = idx%6;
  float acc = b2[a];
  for (int k = 0; k < 128; ++k) acc += (float)u16[(size_t)t*128 + k]*W2[a*128 + k];
  y[idx] = acc;
}

// ---------------- host driver ----------------
extern "C" void kernel_launch(void* const* d_in, const int* in_sizes, int n_in,
                              void* d_out, int out_size, void* d_ws, size_t ws_size,
                              hipStream_t stream){
  (void)in_sizes; (void)n_in; (void)out_size; (void)ws_size;
  const float* x         = (const float*)d_in[0];
  const float* edge_attr = (const float*)d_in[1];
  const float* W0        = (const float*)d_in[2];
  const float* b0        = (const float*)d_in[3];
  const float* We1       = (const float*)d_in[4];
  const float* be1       = (const float*)d_in[5];
  const float* We2       = (const float*)d_in[6];
  const float* be2       = (const float*)d_in[7];
  const float* root      = (const float*)d_in[8];
  const float* conv_b    = (const float*)d_in[9];
  const float* gru_wih   = (const float*)d_in[10];
  const float* gru_whh   = (const float*)d_in[11];
  const float* gru_bih   = (const float*)d_in[12];
  const float* gru_bhh   = (const float*)d_in[13];
  const float* s2s_wi    = (const float*)d_in[14];
  const float* s2s_wh    = (const float*)d_in[15];
  const float* s2s_bi    = (const float*)d_in[16];
  const float* s2s_bh    = (const float*)d_in[17];
  const float* mem_wi    = (const float*)d_in[18];
  const float* mem_bi    = (const float*)d_in[20];
  const float* mem_bh    = (const float*)d_in[21];
  const float* W1        = (const float*)d_in[22];
  const float* b1        = (const float*)d_in[23];
  const float* W2        = (const float*)d_in[24];
  const float* b2        = (const float*)d_in[25];
  const int*   edge_index= (const int*)d_in[26];
  const int*   nonring   = (const int*)d_in[27];

  char* base = (char*)d_ws; size_t off = 0;
  auto alloc = [&](size_t bytes){ void* p = base + off; off = (off + bytes + 255) & ~((size_t)255); return p; };
  float*    out_a  = (float*)alloc((size_t)NNODES*64*4);
  float*    out_b  = (float*)alloc((size_t)NNODES*64*4);
  _Float16* o16_a  = (_Float16*)alloc((size_t)NNODES*64*2);
  _Float16* o16_b  = (_Float16*)alloc((size_t)NNODES*64*2);
  float*    agg    = (float*)alloc((size_t)NNODES*64*4);
  float*    invdeg = (float*)alloc((size_t)NNODES*4);
  float*    hidden = (float*)alloc((size_t)NEDGES*128*4);
  _Float16* hid_sw = (_Float16*)alloc((size_t)NEDGES*128*2);
  _Float16* we2_sw = (_Float16*)alloc((size_t)4096*128*2);
  _Float16* root16 = (_Float16*)alloc((size_t)64*64*2);
  _Float16* wih16  = (_Float16*)alloc((size_t)192*64*2);
  _Float16* whh16  = (_Float16*)alloc((size_t)192*64*2);
  _Float16* mwi16  = (_Float16*)alloc((size_t)1536*384*2);
  _Float16* w116   = (_Float16*)alloc((size_t)128*384*2);
  _Float16* zin16  = (_Float16*)alloc((size_t)NTORS*384*2);
  _Float16* hm16   = (_Float16*)alloc((size_t)NTORS*384*2);
  _Float16* u16    = (_Float16*)alloc((size_t)NTORS*128*2);
  float*    evec   = (float*)alloc((size_t)NNODES*4);
  float*    sm     = (float*)alloc((size_t)1024*4);

  cvt_f16_kernel<<<(64*64+255)/256,256,0,stream>>>(root, root16, 64*64);
  cvt_f16_kernel<<<(192*64+255)/256,256,0,stream>>>(gru_wih, wih16, 192*64);
  cvt_f16_kernel<<<(192*64+255)/256,256,0,stream>>>(gru_whh, whh16, 192*64);
  cvt_f16_kernel<<<(1536*384+255)/256,256,0,stream>>>(mem_wi, mwi16, 1536*384);
  cvt_f16_kernel<<<(128*384+255)/256,256,0,stream>>>(W1, w116, 128*384);

  lin0_kernel<<<(NNODES*64+255)/256,256,0,stream>>>(x, W0, b0, out_a, o16_a);
  edge_hidden_kernel<<<(NEDGES*128+255)/256,256,0,stream>>>(edge_attr, We1, be1, hidden);
  swz_hidden_kernel<<<(NEDGES*128+255)/256,256,0,stream>>>(hidden, hid_sw);
  swz_we2_kernel<<<(4096*128+255)/256,256,0,stream>>>(We2, we2_sw);

  fzero_kernel<<<(NNODES+255)/256,256,0,stream>>>(invdeg, NNODES);
  deg_kernel<<<(NEDGES+255)/256,256,0,stream>>>(edge_index, invdeg);
  invdeg_kernel<<<(NNODES+255)/256,256,0,stream>>>(invdeg);

  float* cur = out_a; float* nxt = out_b;
  _Float16* c16 = o16_a; _Float16* n16 = o16_b;
  for (int it = 0; it < 6; ++it){
    fzero_kernel<<<(NNODES*64+255)/256,256,0,stream>>>(agg, NNODES*64);
    msg_kernel<<<(NTILE_E + WV - 1)/WV, 32*WV, 0, stream>>>(cur, hid_sw, we2_sw, be2, edge_index, agg);
    node_update_kernel<<<(NTILE_N + WV - 1)/WV, 32*WV, 0, stream>>>(
        agg, invdeg, cur, c16, root16, wih16, whh16, conv_b, gru_bih, gru_bhh, nxt, n16);
    float* tf = cur; cur = nxt; nxt = tf;
    _Float16* th = c16; c16 = n16; n16 = th;
  }

  fzero_kernel<<<4,256,0,stream>>>(sm, 1024);
  for (int it = 0; it < 6; ++it){
    s2s_lstm_kernel<<<1,256,0,stream>>>(s2s_wi, s2s_wh, s2s_bi, s2s_bh, sm);
    s2s_scores_kernel<<<(NNODES+255)/256,256,0,stream>>>(cur, evec, sm);
    s2s_accum_kernel<<<(NNODES+255)/256,256,0,stream>>>(cur, evec, sm);
    s2s_final_kernel<<<1,128,0,stream>>>(sm);
  }

  build_zin_kernel<<<(NTORS*384+255)/256,256,0,stream>>>(cur, nonring, sm, zin16);
  mem_lstm_kernel<<<(64*24 + WV - 1)/WV, 32*WV, 0, stream>>>(zin16, mwi16, mem_bi, mem_bh, hm16);
  head_u_kernel<<<(64*8 + WV - 1)/WV, 32*WV, 0, stream>>>(hm16, w116, b1, u16);
  head_y_kernel<<<(NTORS*6+255)/256,256,0,stream>>>(u16, W2, b2, (float*)d_out);
}
